// InfiniAttention_53970559042190
// MI455X (gfx1250) — matrix operand order repaired and checked by reading, not compiled, not run on hardware
//
#include <hip/hip_runtime.h>
#include <hip/hip_bf16.h>

// ---------------------------------------------------------------------------
// InfiniAttention for MI455X (gfx1250), wave32 + v_wmma_f32_16x16x32_f16.
// All WMMA fragments are serviced by two 8-half (2x ds_load_b128, 16 halves apart)
// reads; stationary operands are pre-transposed once so B-fragments are
// K-contiguous in LDS.  Layouts per CDNA5 ISA 7.12.2.
// ---------------------------------------------------------------------------

#define HEADS 16
#define HDIM  128
#define HID   2048
#define SEQ   2048
#define ATT_SCALE 0.08838834764831845f   // 1/sqrt(128)
#define NEG_BIG  -1000000000.0f

typedef __attribute__((ext_vector_type(16))) _Float16 v16h;
typedef __attribute__((ext_vector_type(8)))  _Float16 v8h;
typedef __attribute__((ext_vector_type(8)))  float    v8f;

__device__ __forceinline__ v16h cat8(v8h lo, v8h hi) {
  return __builtin_shufflevector(lo, hi, 0, 1, 2, 3, 4, 5, 6, 7, 8, 9, 10, 11,
                                 12, 13, 14, 15);
}

// A-fragment 16x32 f16 (row-major source, ld halves, 16B-aligned rows):
//   lane m=lane&15, half=lane>>4 : halves [k0+8h .. +8) ++ [k0+16+8h .. +8)
__device__ __forceinline__ v16h frag_a16(const _Float16* base, int ld, int row0,
                                         int k0, int lane) {
  int m = lane & 15, half = lane >> 4;
  const _Float16* p = base + (size_t)(row0 + m) * ld + k0 + 8 * half;
  return cat8(*(const v8h*)p, *(const v8h*)(p + 16));
}

// B-fragment 32x16 f16 from a TRANSPOSED source (rows = n, cols = k):
//   lane n=lane&15, half=lane>>4 : halves [k0+8h .. +8) ++ [k0+16+8h .. +8) of row n0+n
__device__ __forceinline__ v16h frag_bT16(const _Float16* base, int ld, int n0,
                                          int k0, int lane) {
  int n = lane & 15, half = lane >> 4;
  const _Float16* p = base + (size_t)(n0 + n) * ld + k0 + 8 * half;
  return cat8(*(const v8h*)p, *(const v8h*)(p + 16));
}

__device__ __forceinline__ v8f wmma_f16(v16h a, v16h b, v8f c) {
  return __builtin_amdgcn_wmma_f32_16x16x32_f16(false, a, false, b, (short)0, c,
                                                false, false);
}

// ---------------------------------------------------------------------------
// Batched WMMA GEMM: C = A(f16,[M][K]) * BT(f16,[N][K])^T (+bias), f32 out.
// Tile 128x128x32, 256 threads = 8 waves (2x4), 8 WMMA tiles per wave.
// grid = (N/128, M/128, batch)
// ---------------------------------------------------------------------------
__global__ __launch_bounds__(256) void wmma_gemm(
    const _Float16* __restrict__ A, const _Float16* __restrict__ BT,
    const float* __restrict__ bias, float* __restrict__ C, int K, int lda,
    int ldbt, int ldc, long sA, long sB, long sC) {
  __shared__ __align__(16) _Float16 As[128][40];
  __shared__ __align__(16) _Float16 Bs[128][40];  // rows = n, cols = k

  const _Float16* Ab = A + (long)blockIdx.z * sA;
  const _Float16* Bb = BT + (long)blockIdx.z * sB;
  float* Cb = C + (long)blockIdx.z * sC;

  int rowBase = blockIdx.y * 128;
  int colBase = blockIdx.x * 128;

  int t = threadIdx.x;
  int wave = t >> 5, lane = t & 31;
  int wr = wave >> 2, wc = wave & 3;  // 2x4 wave grid
  int n = lane & 15, half = lane >> 4;

  int lr = t >> 1, lc = (t & 1) * 16;  // loader: 128 rows x 32 halves, 2xv8h

  v8f acc[4][2];
#pragma unroll
  for (int i = 0; i < 4; ++i)
#pragma unroll
    for (int j = 0; j < 2; ++j) acc[i][j] = (v8f)(0.0f);

  for (int k0 = 0; k0 < K; k0 += 32) {
    const _Float16* ga = Ab + (size_t)(rowBase + lr) * lda + k0 + lc;
    const _Float16* gb = Bb + (size_t)(colBase + lr) * ldbt + k0 + lc;
    *(v8h*)&As[lr][lc] = *(const v8h*)ga;
    *(v8h*)&As[lr][lc + 8] = *(const v8h*)(ga + 8);
    *(v8h*)&Bs[lr][lc] = *(const v8h*)gb;
    *(v8h*)&Bs[lr][lc + 8] = *(const v8h*)(gb + 8);
    if (k0 + 32 < K) {  // speculative prefetch -> global_prefetch
      __builtin_prefetch(ga + 32, 0, 1);
      __builtin_prefetch(gb + 32, 0, 1);
    }
    __syncthreads();

    v16h bf[2];
#pragma unroll
    for (int j = 0; j < 2; ++j)
      bf[j] = frag_bT16(&Bs[0][0], 40, wc * 32 + j * 16, 0, lane);
#pragma unroll
    for (int i = 0; i < 4; ++i) {
      v16h af = frag_a16(&As[0][0], 40, wr * 64 + i * 16, 0, lane);
#pragma unroll
      for (int j = 0; j < 2; ++j) acc[i][j] = wmma_f16(af, bf[j], acc[i][j]);
    }
    __syncthreads();
  }

#pragma unroll
  for (int i = 0; i < 4; ++i)
#pragma unroll
    for (int j = 0; j < 2; ++j) {
      int col = colBase + wc * 32 + j * 16 + n;
      float bv = bias ? bias[col] : 0.0f;
#pragma unroll
      for (int r = 0; r < 8; ++r) {
        int row = rowBase + wr * 64 + i * 16 + r + 8 * half;
        Cb[(size_t)row * ldc + col] = acc[i][j][r] + bv;
      }
    }
}

// ---------------------------------------------------------------------------
// Conversions / transposes (one-shot, L2-resident)
// ---------------------------------------------------------------------------
__global__ void f32_to_f16(const float* __restrict__ s, _Float16* __restrict__ d,
                           int nElem) {
  int i = blockIdx.x * blockDim.x + threadIdx.x;
  if (i < nElem) d[i] = (_Float16)s[i];
}

// src f32 [R][C] -> dst f16 [C][R]; grid (C/32, R/32, batch), 256 threads
__global__ __launch_bounds__(256) void transpose_f32_f16(
    const float* __restrict__ src, _Float16* __restrict__ dst, int R, int C) {
  __shared__ float tile[32][33];
  size_t zb = (size_t)blockIdx.z * R * C;
  int bx = blockIdx.x * 32, by = blockIdx.y * 32;
  int tx = threadIdx.x & 31, ty = threadIdx.x >> 5;
#pragma unroll
  for (int i = 0; i < 4; ++i)
    tile[ty + i * 8][tx] = src[zb + (size_t)(by + ty + i * 8) * C + bx + tx];
  __syncthreads();
#pragma unroll
  for (int i = 0; i < 4; ++i)
    dst[zb + (size_t)(bx + ty + i * 8) * R + by + tx] =
        (_Float16)tile[tx][ty + i * 8];
}

// src f16 [R][C] -> dst f16 [C][R]
__global__ __launch_bounds__(256) void transpose_f16(
    const _Float16* __restrict__ src, _Float16* __restrict__ dst, int R,
    int C) {
  __shared__ _Float16 tile[32][34];
  int bx = blockIdx.x * 32, by = blockIdx.y * 32;
  int tx = threadIdx.x & 31, ty = threadIdx.x >> 5;
#pragma unroll
  for (int i = 0; i < 4; ++i)
    tile[ty + i * 8][tx] = src[(size_t)(by + ty + i * 8) * C + bx + tx];
  __syncthreads();
#pragma unroll
  for (int i = 0; i < 4; ++i)
    dst[(size_t)(bx + ty + i * 8) * R + by + tx] = tile[tx][ty + i * 8];
}

// ---------------------------------------------------------------------------
// RoPE + ELU feature maps ([S,HID] layout f16 outputs).
// ---------------------------------------------------------------------------
__global__ void rope_kernel(const float* __restrict__ qraw,
                            const float* __restrict__ kraw,
                            const int* __restrict__ pos_ids,
                            _Float16* __restrict__ qr16,
                            _Float16* __restrict__ kr16,
                            _Float16* __restrict__ sq16,
                            _Float16* __restrict__ sk16,
                            _Float16* __restrict__ v16) {
  int idx = blockIdx.x * blockDim.x + threadIdx.x;  // SEQ*HID threads
  int s = idx >> 11, c = idx & (HID - 1), d = c & (HDIM - 1);
  float pos = (float)pos_ids[s];
  int jj = d & 63;
  float inv = __powf(10000.0f, -((float)(2 * jj)) / 128.0f);
  float sn, cs;
  __sincosf(pos * inv, &sn, &cs);
  float q = qraw[idx], k = kraw[idx];
  float qrot = (d < 64) ? -qraw[idx + 64] : qraw[idx - 64];
  float krot = (d < 64) ? -kraw[idx + 64] : kraw[idx - 64];
  float qv = q * cs + qrot * sn;
  float kv = k * cs + krot * sn;
  qr16[idx] = (_Float16)qv;
  kr16[idx] = (_Float16)kv;
  sq16[idx] = (_Float16)(qv > 0.0f ? qv + 1.0f : __expf(qv));  // elu+1
  sk16[idx] = (_Float16)(kv > 0.0f ? kv + 1.0f : __expf(kv));
  v16[idx] = (_Float16)q;  // v = q (pre-RoPE)
}

// mem_den[h][s] = sum_d sq[s,h,d]*z[h,d] + eps
__global__ void den_kernel(const _Float16* __restrict__ sq,
                           const float* __restrict__ z,
                           float* __restrict__ den) {
  int idx = blockIdx.x * blockDim.x + threadIdx.x;  // HEADS*SEQ
  int h = idx >> 11, s = idx & (SEQ - 1);
  float acc = 0.0f;
#pragma unroll 8
  for (int d = 0; d < HDIM; ++d)
    acc += (float)sq[(size_t)s * HID + h * HDIM + d] * z[h * HDIM + d];
  den[idx] = acc + 1e-8f;
}

// z_new[h][d] = z[h][d] + sum_s skT[h*D+d][s]   (contiguous reduction)
__global__ void znew_kernel(const _Float16* __restrict__ skT,
                            const float* __restrict__ z,
                            float* __restrict__ zout) {
  int idx = blockIdx.x * blockDim.x + threadIdx.x;  // HEADS*HDIM = 2048
  float acc = z[idx];
  const _Float16* row = skT + (size_t)idx * SEQ;
  for (int s = 0; s < SEQ; ++s) acc += (float)row[s];
  zout[idx] = acc;
}

// ---------------------------------------------------------------------------
// Flash attention (causal): grid (SEQ/64, HEADS), 128 threads = 4 waves.
// K staged row-major (QK^T B-frag is d-contiguous); V staged from the
// pre-transposed vT so the PV B-frag is t-contiguous.
// ---------------------------------------------------------------------------
__global__ __launch_bounds__(128) void flash_attn(
    const _Float16* __restrict__ qr, const _Float16* __restrict__ kr,
    const _Float16* __restrict__ vT, float* __restrict__ out) {
  __shared__ __align__(16) _Float16 q_s[64][136];
  __shared__ __align__(16) _Float16 k_s[32][136];
  __shared__ __align__(16) _Float16 vs_t[128][40];  // [d][t]
  __shared__ __align__(16) _Float16 p_s[4][16][32];

  int h = blockIdx.y, qb = blockIdx.x;
  int t = threadIdx.x, wave = t >> 5, lane = t & 31;
  int n = lane & 15, half = lane >> 4;
  int qkr = t >> 4, qkc = (t & 15) * 8;  // 8 rows x 128 halves per iter
  int vr = t >> 2, vc = (t & 3) * 8;     // 32 rows x 32 halves per iter

#pragma unroll
  for (int i = 0; i < 8; ++i)
    *(v8h*)&q_s[qkr + i * 8][qkc] =
        *(const v8h*)(qr + (size_t)(qb * 64 + qkr + i * 8) * HID + h * HDIM +
                      qkc);

  v8f acc[8];
#pragma unroll
  for (int j = 0; j < 8; ++j) acc[j] = (v8f)(0.0f);
  float mstat[8], lstat[8];
#pragma unroll
  for (int r = 0; r < 8; ++r) {
    mstat[r] = -3.0e38f;
    lstat[r] = 0.0f;
  }

  int row_base = qb * 64 + wave * 16;
  int tmax = qb * 64 + 64;  // causal
  __syncthreads();

  for (int t0 = 0; t0 < tmax; t0 += 32) {
#pragma unroll
    for (int i = 0; i < 4; ++i) {
      *(v8h*)&k_s[qkr + i * 8][qkc] =
          *(const v8h*)(kr + (size_t)(t0 + qkr + i * 8) * HID + h * HDIM + qkc);
      *(v8h*)&vs_t[vr + i * 32][vc] =
          *(const v8h*)(vT + (size_t)(h * HDIM + vr + i * 32) * SEQ + t0 + vc);
    }
    __syncthreads();

    // scores: 16x32 per wave = 2 WMMA tiles, K over D=128 (4 chunks)
    v8f s0 = (v8f)(0.0f), s1 = (v8f)(0.0f);
#pragma unroll
    for (int kc = 0; kc < 4; ++kc) {
      v16h aq = frag_a16(&q_s[0][0], 136, wave * 16, kc * 32, lane);
      v16h b0 = frag_bT16(&k_s[0][0], 136, 0, kc * 32, lane);
      v16h b1 = frag_bT16(&k_s[0][0], 136, 16, kc * 32, lane);
      s0 = wmma_f16(aq, b0, s0);
      s1 = wmma_f16(aq, b1, s1);
    }

    // online softmax update (row stats live within 16-lane halves)
#pragma unroll
    for (int r = 0; r < 8; ++r) {
      int row_g = row_base + r + 8 * half;
      float x0 = s0[r] * ATT_SCALE + ((t0 + n) > row_g ? NEG_BIG : 0.0f);
      float x1 = s1[r] * ATT_SCALE + ((t0 + 16 + n) > row_g ? NEG_BIG : 0.0f);
      float mx = fmaxf(x0, x1);
#pragma unroll
      for (int m = 1; m < 16; m <<= 1) mx = fmaxf(mx, __shfl_xor(mx, m, 32));
      float mnew = fmaxf(mstat[r], mx);
      float alpha = __expf(mstat[r] - mnew);
      float p0 = __expf(x0 - mnew);
      float p1 = __expf(x1 - mnew);
      float ps = p0 + p1;
#pragma unroll
      for (int m = 1; m < 16; m <<= 1) ps += __shfl_xor(ps, m, 32);
      lstat[r] = lstat[r] * alpha + ps;
      mstat[r] = mnew;
#pragma unroll
      for (int j = 0; j < 8; ++j) acc[j][r] *= alpha;
      p_s[wave][r + 8 * half][n] = (_Float16)p0;  // C-layout -> LDS
      p_s[wave][r + 8 * half][16 + n] = (_Float16)p1;
    }

    // P as A-fragment (same-wave LDS RAW kept in order), then P @ V
    v16h ap = frag_a16(&p_s[wave][0][0], 32, 0, 0, lane);
#pragma unroll
    for (int j = 0; j < 8; ++j) {
      v16h bv = frag_bT16(&vs_t[0][0], 40, j * 16, 0, lane);
      acc[j] = wmma_f16(ap, bv, acc[j]);
    }
    __syncthreads();
  }

#pragma unroll
  for (int j = 0; j < 8; ++j)
#pragma unroll
    for (int r = 0; r < 8; ++r) {
      int row = row_base + r + 8 * half;
      out[(size_t)row * HID + h * HDIM + j * 16 + n] = acc[j][r] / lstat[r];
    }
}

// ---------------------------------------------------------------------------
// M_new[h] = M[h] + sk[h]^T @ v[h]; A/B both from pre-transposed [HID][SEQ].
// grid = HEADS blocks, 256 threads = 8 waves (16 output rows each).
// ---------------------------------------------------------------------------
__global__ __launch_bounds__(256) void mnew_kernel(
    const _Float16* __restrict__ skT, const _Float16* __restrict__ vT,
    const float* __restrict__ Mold, float* __restrict__ Mout) {
  __shared__ __align__(16) _Float16 a_s[128][40];  // [d][s-chunk]
  __shared__ __align__(16) _Float16 b_s[128][40];  // [e][s-chunk]
  int h = blockIdx.x;
  int t = threadIdx.x, wave = t >> 5, lane = t & 31;
  int n = lane & 15, half = lane >> 4;
  int lr = t >> 1, lc = (t & 1) * 16;

  v8f acc[8];
#pragma unroll
  for (int j = 0; j < 8; ++j) acc[j] = (v8f)(0.0f);

  for (int s0 = 0; s0 < SEQ; s0 += 32) {
    const _Float16* ga = skT + (size_t)(h * HDIM + lr) * SEQ + s0 + lc;
    const _Float16* gb = vT + (size_t)(h * HDIM + lr) * SEQ + s0 + lc;
    *(v8h*)&a_s[lr][lc] = *(const v8h*)ga;
    *(v8h*)&a_s[lr][lc + 8] = *(const v8h*)(ga + 8);
    *(v8h*)&b_s[lr][lc] = *(const v8h*)gb;
    *(v8h*)&b_s[lr][lc + 8] = *(const v8h*)(gb + 8);
    __syncthreads();
    v16h af = frag_a16(&a_s[0][0], 40, wave * 16, 0, lane);
#pragma unroll
    for (int j = 0; j < 8; ++j) {
      v16h bf = frag_bT16(&b_s[0][0], 40, j * 16, 0, lane);
      acc[j] = wmma_f16(af, bf, acc[j]);
    }
    __syncthreads();
  }

#pragma unroll
  for (int j = 0; j < 8; ++j)
#pragma unroll
    for (int r = 0; r < 8; ++r) {
      int dd = wave * 16 + r + 8 * half;
      size_t o = (size_t)h * HDIM * HDIM + (size_t)dd * HDIM + j * 16 + n;
      Mout[o] = Mold[o] + acc[j][r];
    }
}

// combined = g*mem_num/den + (1-g)*attn -> f16 for the final GEMM
__global__ void combine_kernel(const float* __restrict__ mem_num,
                               const float* __restrict__ den,
                               const float* __restrict__ attn,
                               const float* __restrict__ beta,
                               _Float16* __restrict__ comb16) {
  int idx = blockIdx.x * blockDim.x + threadIdx.x;  // SEQ*HID
  int s = idx >> 11, c = idx & (HID - 1), h = c >> 7;
  float g = 1.0f / (1.0f + __expf(-beta[0]));
  float mo = mem_num[idx] / den[h * SEQ + s];
  comb16[idx] = (_Float16)(g * mo + (1.0f - g) * attn[idx]);
}

// ---------------------------------------------------------------------------
// Host launcher
// ---------------------------------------------------------------------------
extern "C" void kernel_launch(void* const* d_in, const int* in_sizes, int n_in,
                              void* d_out, int out_size, void* d_ws,
                              size_t ws_size, hipStream_t stream) {
  const float* hs = (const float*)d_in[0];
  (void)d_in[1];  // attention_mask == causal(-1e9); applied analytically
  const int* pos = (const int*)d_in[2];
  const float* Wq = (const float*)d_in[3];
  const float* bq = (const float*)d_in[4];
  const float* Wk = (const float*)d_in[5];
  const float* bk = (const float*)d_in[6];
  const float* Wo = (const float*)d_in[7];
  const float* beta = (const float*)d_in[8];
  const float* Mmat = (const float*)d_in[9];
  const float* z = (const float*)d_in[10];
  (void)in_sizes; (void)n_in; (void)out_size; (void)ws_size;

  float* out_final = (float*)d_out;                    // SEQ*HID
  float* out_M = out_final + (size_t)SEQ * HID;        // H*D*D
  float* out_z = out_M + (size_t)HEADS * HDIM * HDIM;  // H*D

  const size_t NE = (size_t)SEQ * HID;  // 4 Mi elements
  char* ws = (char*)d_ws;
  auto carveH = [&](size_t nElem) {
    _Float16* p = (_Float16*)ws;
    ws += ((nElem * sizeof(_Float16) + 255) & ~(size_t)255);
    return p;
  };
  auto carveF = [&](size_t nElem) {
    float* p = (float*)ws;
    ws += ((nElem * sizeof(float) + 255) & ~(size_t)255);
    return p;
  };

  _Float16* hs16 = carveH(NE);
  _Float16* WqT16 = carveH(NE);   // [N][K]
  _Float16* WkT16 = carveH(NE);
  _Float16* WoT16 = carveH(NE);
  _Float16* MT16 = carveH((size_t)HEADS * HDIM * HDIM);  // per-head [E][D]
  _Float16* qr16 = carveH(NE);
  _Float16* kr16 = carveH(NE);
  _Float16* sq16 = carveH(NE);
  _Float16* sk16 = carveH(NE);
  _Float16* v16 = carveH(NE);
  _Float16* skT16 = carveH(NE);   // [HID][SEQ]
  _Float16* vT16 = carveH(NE);    // [HID][SEQ]
  _Float16* comb16 = carveH(NE);
  float* qraw = carveF(NE);
  float* kraw = carveF(NE);
  float* mem_num = carveF(NE);
  float* attn = carveF(NE);
  float* den = carveF((size_t)HEADS * SEQ);

  const int CB = 256;
  int nblk = (int)(NE / CB);
  f32_to_f16<<<nblk, CB, 0, stream>>>(hs, hs16, (int)NE);

  dim3 gT(HID / 32, HID / 32, 1);
  transpose_f32_f16<<<gT, 256, 0, stream>>>(Wq, WqT16, HID, HID);
  transpose_f32_f16<<<gT, 256, 0, stream>>>(Wk, WkT16, HID, HID);
  transpose_f32_f16<<<gT, 256, 0, stream>>>(Wo, WoT16, HID, HID);
  transpose_f32_f16<<<dim3(HDIM / 32, HDIM / 32, HEADS), 256, 0, stream>>>(
      Mmat, MT16, HDIM, HDIM);

  // Q = hs @ Wq + bq ; K = hs @ Wk + bk
  dim3 gFull(HID / 128, SEQ / 128, 1);
  wmma_gemm<<<gFull, 256, 0, stream>>>(hs16, WqT16, bq, qraw, HID, HID, HID,
                                       HID, 0, 0, 0);
  wmma_gemm<<<gFull, 256, 0, stream>>>(hs16, WkT16, bk, kraw, HID, HID, HID,
                                       HID, 0, 0, 0);

  rope_kernel<<<nblk, CB, 0, stream>>>(qraw, kraw, pos, qr16, kr16, sq16, sk16,
                                       v16);

  transpose_f16<<<gT, 256, 0, stream>>>(sk16, skT16, SEQ, HID);
  transpose_f16<<<gT, 256, 0, stream>>>(v16, vT16, SEQ, HID);

  den_kernel<<<(HEADS * SEQ) / CB, CB, 0, stream>>>(sq16, z, den);
  znew_kernel<<<(HEADS * HDIM) / CB, CB, 0, stream>>>(skT16, z, out_z);

  // mem_num[h] = sq[h] @ M[h]  (batched over heads; A/C are column slices)
  dim3 gMem(HDIM / 128, SEQ / 128, HEADS);
  wmma_gemm<<<gMem, 256, 0, stream>>>(sq16, MT16, nullptr, mem_num, HDIM, HID,
                                      HDIM, HID, (long)HDIM,
                                      (long)HDIM * HDIM, (long)HDIM);

  flash_attn<<<dim3(SEQ / 64, HEADS), 128, 0, stream>>>(qr16, kr16, vT16, attn);

  mnew_kernel<<<HEADS, 256, 0, stream>>>(skT16, vT16, Mmat, out_M);

  combine_kernel<<<nblk, CB, 0, stream>>>(mem_num, den, attn, beta, comb16);

  // final = combined @ Wo
  wmma_gemm<<<gFull, 256, 0, stream>>>(comb16, WoT16, nullptr, out_final, HID,
                                       HID, HID, HID, 0, 0, 0);
}
